// PoseDescriptorModel_23931557773745
// MI455X (gfx1250) — matrix operand order repaired and checked by reading, not compiled, not run on hardware
//
#include <hip/hip_runtime.h>

typedef __attribute__((ext_vector_type(16))) _Float16 v16h;
typedef __attribute__((ext_vector_type(8)))  float    v8f;

#define FH 60
#define FW 80
#define CCH 256
#define NPIX (FH * FW)
#define PTS_PER_BLK 128
#define FSTRIDE 272          // 256 + 16 halves of padding (anti bank-alignment)
#define SAMP_DW 8            // per-point LDS record: 4 idx + 4 weights (dwords)
#define SAMP_BYTES (PTS_PER_BLK * SAMP_DW * 4)
#define LDS_BYTES (SAMP_BYTES + PTS_PER_BLK * FSTRIDE * 2)

// d_ws layout:
//   [0, 131072)             : Wt   half[256][256], Wt[k][n] = W_lin[n][k]
//   [131072, 131072+2457600): featT half[4800][256], featT[y*80+x][c]
#define WS_WT_OFF   0
#define WS_FEAT_OFF 131072

// ---------------- prep: W_lin (f32, [n][k]) -> Wt (f16, K-major [k][n]) ----------------
__global__ void prep_w_kernel(const float* __restrict__ Wlin, _Float16* __restrict__ Wt) {
    int k = threadIdx.x;     // 0..255 (coalesced read of W_lin row n)
    int n = blockIdx.x;      // 0..255
    Wt[k * CCH + n] = (_Float16)Wlin[n * CCH + k];
}

// ---------------- prep: featmap (f32, C,H,W) -> featT (f16, HW,C) ----------------
__global__ void prep_feat_kernel(const float* __restrict__ fm, _Float16* __restrict__ ft) {
    int pix = blockIdx.x;    // 0..4799
    int c   = threadIdx.x;   // 0..255
    ft[pix * CCH + c] = (_Float16)fm[c * NPIX + pix];
}

// ---------------- fused geometry + bilinear + WMMA GEMM ----------------
__global__ __launch_bounds__(256, 1)
void pose_desc_kernel(const float* __restrict__ vtx,
                      const float* __restrict__ q4,  const float* __restrict__ t3,
                      const float* __restrict__ qc4, const float* __restrict__ tc3,
                      const float* __restrict__ camf, const float* __restrict__ camc,
                      const float* __restrict__ blin,
                      const _Float16* __restrict__ Wt,
                      const _Float16* __restrict__ featT,
                      const int* __restrict__ psPtr,
                      float* __restrict__ out,
                      int nPts)
{
    extern __shared__ char smem[];
    int*      sampI = (int*)smem;
    float*    sampF = (float*)smem;
    _Float16* sfe   = (_Float16*)(smem + SAMP_BYTES);

    const int tid   = threadIdx.x;
    const int pBase = blockIdx.x * PTS_PER_BLK;

    // ---- stage 1: geometry + projection for 128 points ----
    if (tid < PTS_PER_BLK) {
        const int p = pBase + tid;
        float wa = 0.f, wb = 0.f, wc = 0.f, wd = 0.f;
        int ia = 0, ib = 0, ic = 0, id_ = 0;
        if (p < nPts) {
            const float invPs = 1.0f / (float)psPtr[0];
            float vx = vtx[p * 3 + 0], vy = vtx[p * 3 + 1], vz = vtx[p * 3 + 2];
            // rotate by object quaternion q
            float qx = q4[0], qy = q4[1], qz = q4[2], qw = q4[3];
            float ux = qy * vz - qz * vy;
            float uy = qz * vx - qx * vz;
            float uz = qx * vy - qy * vx;
            float cxx = qy * uz - qz * uy;
            float cyy = qz * ux - qx * uz;
            float czz = qx * uy - qy * ux;
            float ox = vx + 2.f * (qw * ux + cxx) + t3[0];
            float oy = vy + 2.f * (qw * uy + cyy) + t3[1];
            float oz = vz + 2.f * (qw * uz + czz) + t3[2];
            // rotate by camera quaternion
            float rx = qc4[0], ry = qc4[1], rz = qc4[2], rw = qc4[3];
            float u2x = ry * oz - rz * oy;
            float u2y = rz * ox - rx * oz;
            float u2z = rx * oy - ry * ox;
            float c2x = ry * u2z - rz * u2y;
            float c2y = rz * u2x - rx * u2z;
            float c2z = rx * u2y - ry * u2x;
            float mx = ox + 2.f * (rw * u2x + c2x) + tc3[0];
            float my = oy + 2.f * (rw * u2y + c2y) + tc3[1];
            float mz = oz + 2.f * (rw * u2z + c2z) + tc3[2];
            float inv = 1.0f / mz;
            float X = ((mx * inv) * camf[0] + camc[0]) * invPs;
            float Y = ((my * inv) * camf[1] + camc[1]) * invPs;
            float fx0 = floorf(X), fy0 = floorf(Y);
            int x0 = (int)fx0, y0 = (int)fy0;
            int x1 = x0 + 1,   y1 = y0 + 1;
            x0 = min(max(x0, 0), FW - 1);
            x1 = min(max(x1, 0), FW - 1);
            y0 = min(max(y0, 0), FH - 1);
            y1 = min(max(y1, 0), FH - 1);
            float X0 = (float)x0, X1 = (float)x1, Y0 = (float)y0, Y1 = (float)y1;
            wa = (X1 - X) * (Y1 - Y);   // (y0,x0)
            wb = (X1 - X) * (Y - Y0);   // (y1,x0)
            wc = (X - X0) * (Y1 - Y);   // (y0,x1)
            wd = (X - X0) * (Y - Y0);   // (y1,x1)
            ia  = y0 * FW + x0;
            ib  = y1 * FW + x0;
            ic  = y0 * FW + x1;
            id_ = y1 * FW + x1;
        }
        int* sp = sampI + tid * SAMP_DW;
        sp[0] = ia; sp[1] = ib; sp[2] = ic; sp[3] = id_;
        float* spf = sampF + tid * SAMP_DW;
        spf[4] = wa; spf[5] = wb; spf[6] = wc; spf[7] = wd;
    }
    __syncthreads();

    // ---- stage 2: cooperative bilinear gather into the LDS A-tile (f16) ----
    {
        const int grp = tid >> 6;          // 4 points processed per iteration
        const int c4  = (tid & 63) * 4;    // 4 channels per thread
        for (int pb = 0; pb < PTS_PER_BLK; pb += 4) {
            const int lp = pb + grp;
            const int*   sp  = sampI + lp * SAMP_DW;
            const float* spf = sampF + lp * SAMP_DW;
            const int ba = sp[0] * CCH, bb = sp[1] * CCH, bc = sp[2] * CCH, bd = sp[3] * CCH;
            const float wa = spf[4], wb = spf[5], wc = spf[6], wd = spf[7];
            union { uint2 u; _Float16 h[4]; } A, B, C, D, R;
            A.u = *(const uint2*)(featT + ba + c4);
            B.u = *(const uint2*)(featT + bb + c4);
            C.u = *(const uint2*)(featT + bc + c4);
            D.u = *(const uint2*)(featT + bd + c4);
#pragma unroll
            for (int j = 0; j < 4; ++j)
                R.h[j] = (_Float16)(wa * (float)A.h[j] + wb * (float)B.h[j] +
                                    wc * (float)C.h[j] + wd * (float)D.h[j]);
            *(uint2*)(sfe + lp * FSTRIDE + c4) = R.u;
        }
    }
    __syncthreads();

    // ---- stage 3: WMMA GEMM: out[128x256] = feats(128x256,f16) x Wt(K-major,f16) ----
    // Wave w owns row tile w. A panel (K=256 -> 8 x v16h = 64 VGPRs) lives in
    // registers for the whole sweep; B is streamed from the L2-resident Wt into
    // per-kStep registers so all 128 strided f16 loads issue before the WMMAs per tile.
    const int wave   = tid >> 5;   // 0..7  == row tile
    const int lane   = tid & 31;
    const int laneLo = lane & 15;
    const int hi     = lane >> 4;  // 0/1: upper K/M half per ISA lane maps
    const int arow   = wave * 16 + laneLo;

    v16h Apan[8];
#pragma unroll
    for (int s = 0; s < 8; ++s) {
        // 16-bit A 16x32 lane map: lane<16 -> K {kb..kb+7, kb+16..kb+23},
        //                          lane>=16 -> shifted by 8
        const int kb = s * 32 + hi * 8;
        union { uint4 u[2]; v16h h; } Af;
        Af.u[0] = *(const uint4*)(sfe + arow * FSTRIDE + kb);
        Af.u[1] = *(const uint4*)(sfe + arow * FSTRIDE + kb + 16);
        Apan[s] = Af.h;
    }

    const _Float16* wrow = Wt + hi * 8 * CCH + laneLo;   // lane: N = laneLo, K base 8*hi
    const bool full = (pBase + PTS_PER_BLK) <= nPts;
    const int rowTop = pBase + wave * 16 + hi * 8;

    if (full) {
#pragma unroll 1
        for (int ct = 0; ct < 16; ++ct) {
            const int colBase = ct * 16;
            v16h Bf[8];
#pragma unroll
            for (int s = 0; s < 8; ++s)
#pragma unroll
                for (int e = 0; e < 16; ++e)
                    Bf[s][e] = wrow[(s * 32 + e + (e >= 8 ? 8 : 0)) * CCH + colBase];
            const float bias = blin[colBase + laneLo];
            v8f acc = {};
#pragma unroll
            for (int s = 0; s < 8; ++s)
                acc = __builtin_amdgcn_wmma_f32_16x16x32_f16(
                        false, Apan[s], false, Bf[s], (short)0, acc, false, false);
            // C/D map: VGPR r -> M = r + 8*hi, N = laneLo
            float* op = out + rowTop * CCH + colBase + laneLo;
#pragma unroll
            for (int r = 0; r < 8; ++r)
                op[r * CCH] = acc[r] + bias;
        }
    } else {
#pragma unroll 1
        for (int ct = 0; ct < 16; ++ct) {
            const int colBase = ct * 16;
            v16h Bf[8];
#pragma unroll
            for (int s = 0; s < 8; ++s)
#pragma unroll
                for (int e = 0; e < 16; ++e)
                    Bf[s][e] = wrow[(s * 32 + e + (e >= 8 ? 8 : 0)) * CCH + colBase];
            const float bias = blin[colBase + laneLo];
            v8f acc = {};
#pragma unroll
            for (int s = 0; s < 8; ++s)
                acc = __builtin_amdgcn_wmma_f32_16x16x32_f16(
                        false, Apan[s], false, Bf[s], (short)0, acc, false, false);
            float* op = out + rowTop * CCH + colBase + laneLo;
#pragma unroll
            for (int r = 0; r < 8; ++r)
                if (rowTop + r < nPts)
                    op[r * CCH] = acc[r] + bias;
        }
    }
}

extern "C" void kernel_launch(void* const* d_in, const int* in_sizes, int n_in,
                              void* d_out, int out_size, void* d_ws, size_t ws_size,
                              hipStream_t stream) {
    const float* vtx  = (const float*)d_in[0];
    const float* q    = (const float*)d_in[1];
    const float* t    = (const float*)d_in[2];
    const float* qc   = (const float*)d_in[3];
    const float* tc   = (const float*)d_in[4];
    const float* camf = (const float*)d_in[5];
    const float* camc = (const float*)d_in[6];
    const float* fm   = (const float*)d_in[7];
    const float* Wl   = (const float*)d_in[8];
    const float* bl   = (const float*)d_in[9];
    const int*   ps   = (const int*)d_in[10];

    _Float16* Wt    = (_Float16*)((char*)d_ws + WS_WT_OFF);
    _Float16* featT = (_Float16*)((char*)d_ws + WS_FEAT_OFF);
    float*    out   = (float*)d_out;

    const int nPts = in_sizes[0] / 3;

    prep_w_kernel<<<CCH, CCH, 0, stream>>>(Wl, Wt);
    prep_feat_kernel<<<NPIX, CCH, 0, stream>>>(fm, featT);

    const int nBlk = (nPts + PTS_PER_BLK - 1) / PTS_PER_BLK;
    pose_desc_kernel<<<nBlk, 256, LDS_BYTES, stream>>>(
        vtx, q, t, qc, tc, camf, camc, bl, Wt, featT, ps, out, nPts);
}
